// Network_15083925144426
// MI455X (gfx1250) — matrix operand order repaired and checked by reading, not compiled, not run on hardware
//
#include <hip/hip_runtime.h>
#include <hip/hip_bf16.h>

typedef __attribute__((ext_vector_type(16))) _Float16 v16h;
typedef __attribute__((ext_vector_type(8)))  float    v8f;

#define EPSBN 1e-5f

// ---------------------------------------------------------------------------
// CDNA5 V_WMMA_F32_16X16X32_F16, wave32. Per-lane layouts used here (A/B k contiguous, not ISA table order):
//   A (16x32 f16): lane L -> M=L&15, kb=(L<16)?0:16; halfs = A[M][kb+0..15]
//                  -> stored PRE-SWIZZLED so each lane's 16
//                  halfs are contiguous (one 32B load).
//   B (32x16 f16): lane L -> N=L&15, kb=(L<16)?0:16; half[j]=B[kb+j][N]
//                  -> activations staged K-major ([point][Kpad] f16) so each
//                  lane's 16 halfs are contiguous (one 32B load).
//   C/D (16x16 f32): lane L -> N=L&15, mb=(L<16)?0:8, reg r = row mb+r.
// ---------------------------------------------------------------------------
#define WMMA_F16(a, b, c) \
  __builtin_amdgcn_wmma_f32_16x16x32_f16(false, (a), false, (b), (short)0, (c), false, false)

// ---------------------------------------------------------------------------
// Setup: fold BN into weights, emit pre-swizzled f16 WMMA A-tiles.
// ---------------------------------------------------------------------------
struct SetupParams {
  const float *gpe_be, *gpe_g, *gpe_m, *gpe_v, *gpe_w;        // 4x10
  const float *lpe1_be, *lpe1_g, *lpe1_m, *lpe1_v, *lpe1_w;   // 8x10
  const float *lpe2_be, *lpe2_g, *lpe2_m, *lpe2_v, *lpe2_w;   // 4x8
  const float *mlp2_be, *mlp2_g, *mlp2_m, *mlp2_v, *mlp2_w;   // 128x64
  const float *qk1_b, *qk1_w;                                 // 64x8
  const float *qk2_b, *qk2_w;                                 // 64x64
  const float *sc_be, *sc_g, *sc_m, *sc_v, *sc_w;             // 128x8
  const float *v1_be, *v1_g, *v1_m, *v1_v, *v1_w;             // 32x8
  const float *v2_be, *v2_g, *v2_m, *v2_v, *v2_w;             // 32x64
  _Float16 *A1swz;  // 6 mtiles x 1 ktile  (qk1 rows 0..63, v1 rows 64..95, K 8->32 pad)
  _Float16 *A2swz;  // 6 mtiles x 2 ktiles (qk2, v2)
  _Float16 *A3swz;  // 8 mtiles x 3 ktiles (mlp2 cols 0..63, sc cols 64..71, pad ->96)
  float *bias1, *bias2, *bias3;
  float *Wl1, *bl1, *Wl2, *bl2, *Wg, *bg;  // BN-folded PE-MLP weights (f32)
};

__device__ inline float bnscale(const float* g, const float* v, int o) {
  return g[o] * rsqrtf(v[o] + EPSBN);
}

struct FoldA1 {
  const SetupParams* p;
  __device__ float operator()(int m, int k) const {
    if (k >= 8) return 0.f;
    if (m < 64) return p->qk1_w[m * 8 + k];
    int o = m - 64;
    return p->v1_w[o * 8 + k] * bnscale(p->v1_g, p->v1_v, o);
  }
};
struct FoldA2 {
  const SetupParams* p;
  __device__ float operator()(int m, int k) const {
    if (m < 64) return p->qk2_w[m * 64 + k];
    int o = m - 64;
    return p->v2_w[o * 64 + k] * bnscale(p->v2_g, p->v2_v, o);
  }
};
struct FoldA3 {
  const SetupParams* p;
  __device__ float operator()(int m, int k) const {
    if (k < 64) return p->mlp2_w[m * 64 + k] * bnscale(p->mlp2_g, p->mlp2_v, m);
    if (k < 72) return p->sc_w[m * 8 + (k - 64)] * bnscale(p->sc_g, p->sc_v, m);
    return 0.f;
  }
};

// Write pre-swizzled A: dst[((mt*KT+kt)*32 + L)*16 + j] = fold(m, k) with
// m = mt*16 + (L&15); k = kt*32 + kb + j, kb = (L&16)?16:0.
template <typename F>
__device__ void write_swz(_Float16* dst, int Mtiles, int KT, F f, int t) {
  const int total = Mtiles * KT * 32 * 16;
  for (int i = t; i < total; i += 256) {
    int j = i & 15;
    int L = (i >> 4) & 31;
    int tile = i >> 9;
    int kt = tile % KT, mt = tile / KT;
    int m = mt * 16 + (L & 15);
    int kb = (L & 16) ? 16 : 0;
    int k = kt * 32 + kb + j;
    dst[i] = (_Float16)f(m, k);
  }
}

__global__ void __launch_bounds__(256) setup_kernel(SetupParams p) {
  const int t = threadIdx.x;
  write_swz(p.A1swz, 6, 1, FoldA1{&p}, t);
  write_swz(p.A2swz, 6, 2, FoldA2{&p}, t);
  write_swz(p.A3swz, 8, 3, FoldA3{&p}, t);

  for (int o = t; o < 64; o += 256) p.bias1[o] = p.qk1_b[o];
  for (int o = t; o < 32; o += 256) {
    float s = bnscale(p.v1_g, p.v1_v, o);
    p.bias1[64 + o] = p.v1_be[o] - p.v1_m[o] * s;
  }
  for (int o = t; o < 64; o += 256) p.bias2[o] = p.qk2_b[o];
  for (int o = t; o < 32; o += 256) {
    float s = bnscale(p.v2_g, p.v2_v, o);
    p.bias2[64 + o] = p.v2_be[o] - p.v2_m[o] * s;
  }
  for (int o = t; o < 128; o += 256) {
    float sm = bnscale(p.mlp2_g, p.mlp2_v, o);
    float ss = bnscale(p.sc_g, p.sc_v, o);
    p.bias3[o] = (p.mlp2_be[o] - p.mlp2_m[o] * sm) + (p.sc_be[o] - p.sc_m[o] * ss);
  }

  // BN-folded PE MLPs (f32; tiny K, consumed by per-thread matvecs)
  for (int i = t; i < 8 * 10; i += 256)
    p.Wl1[i] = p.lpe1_w[i] * bnscale(p.lpe1_g, p.lpe1_v, i / 10);
  for (int o = t; o < 8; o += 256) {
    float s = bnscale(p.lpe1_g, p.lpe1_v, o);
    p.bl1[o] = p.lpe1_be[o] - p.lpe1_m[o] * s;
  }
  for (int i = t; i < 4 * 8; i += 256)
    p.Wl2[i] = p.lpe2_w[i] * bnscale(p.lpe2_g, p.lpe2_v, i >> 3);
  for (int o = t; o < 4; o += 256) {
    float s = bnscale(p.lpe2_g, p.lpe2_v, o);
    p.bl2[o] = p.lpe2_be[o] - p.lpe2_m[o] * s;
  }
  for (int i = t; i < 4 * 10; i += 256)
    p.Wg[i] = p.gpe_w[i] * bnscale(p.gpe_g, p.gpe_v, i / 10);
  for (int o = t; o < 4; o += 256) {
    float s = bnscale(p.gpe_g, p.gpe_v, o);
    p.bg[o] = p.gpe_be[o] - p.gpe_m[o] * s;
  }
}

__global__ void __launch_bounds__(256) zero_f32_kernel(float* p, int count) {
  int i = blockIdx.x * 256 + threadIdx.x;
  if (i < count) p[i] = 0.f;
}

// Pack feature into K-major f16 staging: featT[b][col][0..7]=feature, 8..31=0;
// f2featT[b][col][64..71]=feature, 72..95=0 (cols 0..63 filled by gt_kernel).
__global__ void __launch_bounds__(256)
pack_feature_kernel(const float* __restrict__ feature, _Float16* __restrict__ featT,
                    _Float16* __restrict__ f2featT, int n) {
  const int idx = blockIdx.x * 256 + threadIdx.x;  // b*n + col
  const int b = idx / n, col = idx - b * n;
  const float* fb = feature + (size_t)b * 8 * n;
  _Float16* r1 = featT + (size_t)idx * 32;
  _Float16* r2 = f2featT + (size_t)idx * 96 + 64;
#pragma unroll
  for (int c = 0; c < 8; ++c) {
    _Float16 h = (_Float16)fb[(size_t)c * n + col];
    r1[c] = h;
    r2[c] = h;
  }
#pragma unroll
  for (int c = 8; c < 32; ++c) r1[c] = (_Float16)0.f;
#pragma unroll
  for (int c = 8; c < 32; ++c) r2[c] = (_Float16)0.f;
}

// ---------------------------------------------------------------------------
// WMMA GEMM, fully unrolled over K. A pre-swizzled f16, B K-major f16.
// One wave -> one 16x16 output tile; 8 waves/block. Epilogue: bias + (leaky)ReLU.
// POINT_MAJOR=true : C[point][Mld] (contiguous 8-float -> 2x b128 per lane)
// POINT_MAJOR=false: C[m][n]       (reference output layout for d_out)
// ---------------------------------------------------------------------------
template <int KP, bool LEAKY, bool POINT_MAJOR>
__global__ void __launch_bounds__(256)
wmma_gemm_kernel(const _Float16* __restrict__ Aswz, const float* __restrict__ bias,
                 const _Float16* __restrict__ Bt, float* __restrict__ C,
                 int n, int Mld, long long strideBt, long long strideC) {
  constexpr int KT = KP / 32;
  const int lane = threadIdx.x & 31;
  const int wave = threadIdx.x >> 5;
  const int n0 = (blockIdx.x * 8 + wave) * 16;
  const int mt = blockIdx.y;
  const _Float16* Bb = Bt + (long long)blockIdx.z * strideBt;
  float* Cb = C + (long long)blockIdx.z * strideC;
  const int ncol = n0 + (lane & 15);
  const int kb = (lane & 16) ? 16 : 0;
  const _Float16* brow = Bb + (size_t)ncol * KP + kb;
  const _Float16* arow = Aswz + ((size_t)(mt * KT) * 32 + lane) * 16;

  v8f acc = {};
#pragma unroll
  for (int kt = 0; kt < KT; ++kt) {
    v16h a = *(const v16h*)(arow + (size_t)kt * 32 * 16);
    v16h b = *(const v16h*)(brow + kt * 32);
    acc = WMMA_F16(a, b, acc);
  }
  const int mb = mt * 16 + ((lane & 16) ? 8 : 0);
  float e[8];
#pragma unroll
  for (int r = 0; r < 8; ++r) {
    float x = acc[r] + bias[mb + r];
    e[r] = LEAKY ? (x > 0.f ? x : 0.2f * x) : fmaxf(x, 0.f);
  }
  if (POINT_MAJOR) {
    float* dst = Cb + (size_t)ncol * Mld + mb;  // 32B aligned (mb mult of 8)
    *(float4*)(dst)     = make_float4(e[0], e[1], e[2], e[3]);
    *(float4*)(dst + 4) = make_float4(e[4], e[5], e[6], e[7]);
  } else {
#pragma unroll
    for (int r = 0; r < 8; ++r) Cb[(size_t)(mb + r) * n + ncol] = e[r];
  }
}

// ---------------------------------------------------------------------------
// GTL module: one thread per (b, g, point). q1v1 POINT-MAJOR [b*n][96]:
// cols 0..63 query, 64..95 value. All gathers are contiguous float4 chunks.
// Writes f1t K-major f16 for the next WMMA GEMM; centrality via f32 atomics.
// ---------------------------------------------------------------------------
__global__ void __launch_bounds__(256)
gtl_kernel(const float* __restrict__ q1v1, const float* __restrict__ xyz,
           const int* __restrict__ nidx, const float* __restrict__ Wl1,
           const float* __restrict__ bl1, _Float16* __restrict__ f1t,
           float* __restrict__ ac, int n, int k) {
  const int tid = blockIdx.x * 256 + threadIdx.x;
  const int bg = tid / n, col = tid - bg * n;
  const int b = bg >> 2, g = bg & 3;
  if (k > 16) k = 16;
  const float* P = q1v1 + (size_t)(b * n + col) * 96;

  float lq[16];
  {
    const float4* q4 = (const float4*)(P + g * 16);
#pragma unroll
    for (int u = 0; u < 4; ++u) {
      float4 t = q4[u];
      lq[4 * u + 0] = t.x; lq[4 * u + 1] = t.y; lq[4 * u + 2] = t.z; lq[4 * u + 3] = t.w;
    }
  }
  const float xn0 = xyz[(size_t)(b * n + col) * 3 + 0];
  const float xn1 = xyz[(size_t)(b * n + col) * 3 + 1];
  const float xn2 = xyz[(size_t)(b * n + col) * 3 + 2];
  const int* idp = nidx + (size_t)(b * n + col) * k;

  float dots[16];
  float mx = -3.4e38f;
  for (int j = 0; j < k; ++j) {
    int nb = idp[j];
    const float4* k4 = (const float4*)(q1v1 + (size_t)(b * n + nb) * 96 + g * 16);
    float d = 0.f;
#pragma unroll
    for (int u = 0; u < 4; ++u) {
      float4 t = k4[u];
      d += lq[4 * u + 0] * t.x + lq[4 * u + 1] * t.y + lq[4 * u + 2] * t.z + lq[4 * u + 3] * t.w;
    }
    dots[j] = d;
    mx = fmaxf(mx, d);
  }
  float ssum = 0.f;
  for (int j = 0; j < k; ++j) { dots[j] = __expf(dots[j] - mx); ssum += dots[j]; }
  const float inv = 1.f / ssum;

  float outv[16];
#pragma unroll
  for (int c = 0; c < 16; ++c) outv[c] = 0.f;

  for (int j = 0; j < k; ++j) {
    int nb = idp[j];
    float la = dots[j] * inv;
    const float4* v4 = (const float4*)(q1v1 + (size_t)(b * n + nb) * 96 + 64 + g * 8);
#pragma unroll
    for (int u = 0; u < 2; ++u) {
      float4 t = v4[u];
      outv[4 * u + 0] += la * t.x; outv[4 * u + 1] += la * t.y;
      outv[4 * u + 2] += la * t.z; outv[4 * u + 3] += la * t.w;
    }
    float p0 = xyz[(size_t)(b * n + nb) * 3 + 0];
    float p1 = xyz[(size_t)(b * n + nb) * 3 + 1];
    float p2 = xyz[(size_t)(b * n + nb) * 3 + 2];
    float r0 = xn0 - p0, r1 = xn1 - p1, r2 = xn2 - p2;
    float dist = sqrtf(r0 * r0 + r1 * r1 + r2 * r2);
    float rp[10] = {dist, r0, r1, r2, xn0, xn1, xn2, p0, p1, p2};
#pragma unroll
    for (int c = 0; c < 8; ++c) {
      float s = bl1[c];
#pragma unroll
      for (int t = 0; t < 10; ++t) s += Wl1[c * 10 + t] * rp[t];
      outv[8 + c] += la * fmaxf(s, 0.f);
    }
    atomicAdd(&ac[(size_t)bg * n + nb], la);  // centrality scatter-add
  }
  _Float16* Ft = f1t + ((size_t)(b * n + col)) * 64 + g * 16;
#pragma unroll
  for (int c = 0; c < 16; ++c) Ft[c] = (_Float16)outv[c];
}

// ---------------------------------------------------------------------------
// Top-16 of ac[bg][0..n-1] by iterative block-wide argmax (order irrelevant:
// downstream softmax+sum over k is permutation-invariant).
// ---------------------------------------------------------------------------
__global__ void __launch_bounds__(256)
topk_kernel(const float* __restrict__ ac, float* __restrict__ scratch,
            int* __restrict__ gidx, int n) {
  const int row = blockIdx.x;
  const float* src = ac + (size_t)row * n;
  float* buf = scratch + (size_t)row * n;
  for (int i = threadIdx.x; i < n; i += 256) buf[i] = src[i];
  __syncthreads();
  __shared__ float sval[256];
  __shared__ int sidx[256];
  for (int t = 0; t < 16; ++t) {
    float best = -3.4e38f;
    int bi = 0x7fffffff;
    for (int i = threadIdx.x; i < n; i += 256) {
      float v = buf[i];
      if (v > best || (v == best && i < bi)) { best = v; bi = i; }
    }
    sval[threadIdx.x] = best;
    sidx[threadIdx.x] = bi;
    __syncthreads();
    for (int s = 128; s > 0; s >>= 1) {
      if (threadIdx.x < s) {
        float ov = sval[threadIdx.x + s];
        int oi = sidx[threadIdx.x + s];
        if (ov > sval[threadIdx.x] || (ov == sval[threadIdx.x] && oi < sidx[threadIdx.x])) {
          sval[threadIdx.x] = ov;
          sidx[threadIdx.x] = oi;
        }
      }
      __syncthreads();
    }
    if (threadIdx.x == 0) {
      gidx[row * 16 + t] = sidx[0];
      buf[sidx[0]] = -3.4e38f;
    }
    __syncthreads();
  }
}

// ---------------------------------------------------------------------------
// GT module: one thread per (b, g, point). q2v2 POINT-MAJOR [b*n][96]:
// cols 0..31 local q, 32..63 global q, 64..79 local v, 80..95 global v.
// Global keys/values/xyz of the 16 central points cached in LDS per block
// (a block shares one (b,g)). fx (lpe1 output) recomputed on the fly.
// Writes f2featT cols 0..63 (K-major f16) for the final WMMA GEMM.
// ---------------------------------------------------------------------------
__global__ void __launch_bounds__(256)
gt_kernel(const float* __restrict__ q2v2, const float* __restrict__ xyz,
          const int* __restrict__ nidx, const int* __restrict__ gidx,
          const float* __restrict__ Wl1, const float* __restrict__ bl1,
          const float* __restrict__ Wl2, const float* __restrict__ bl2,
          const float* __restrict__ Wg, const float* __restrict__ bg_,
          _Float16* __restrict__ f2featT, int n, int k) {
  const int tid = blockIdx.x * 256 + threadIdx.x;
  const int bg = tid / n, col = tid - bg * n;
  const int b = bg >> 2, g = bg & 3;
  if (k > 16) k = 16;

  __shared__ float sgxyz[16 * 3];
  __shared__ float sgk[16 * 8];
  __shared__ float sgvn[16 * 4];
  if (threadIdx.x < 16) {
    int gid = gidx[bg * 16 + threadIdx.x];
    const float* R = q2v2 + (size_t)(b * n + gid) * 96;
    sgxyz[threadIdx.x * 3 + 0] = xyz[(size_t)(b * n + gid) * 3 + 0];
    sgxyz[threadIdx.x * 3 + 1] = xyz[(size_t)(b * n + gid) * 3 + 1];
    sgxyz[threadIdx.x * 3 + 2] = xyz[(size_t)(b * n + gid) * 3 + 2];
    for (int c = 0; c < 8; ++c) sgk[threadIdx.x * 8 + c] = R[32 + g * 8 + c];
    for (int c = 0; c < 4; ++c) sgvn[threadIdx.x * 4 + c] = R[80 + g * 4 + c];
  }
  __syncthreads();

  const float* P = q2v2 + (size_t)(b * n + col) * 96;
  const float xn0 = xyz[(size_t)(b * n + col) * 3 + 0];
  const float xn1 = xyz[(size_t)(b * n + col) * 3 + 1];
  const float xn2 = xyz[(size_t)(b * n + col) * 3 + 2];

  // ---- local branch ----
  float lq[8];
  {
    const float4* q4 = (const float4*)(P + g * 8);
#pragma unroll
    for (int u = 0; u < 2; ++u) {
      float4 t = q4[u];
      lq[4 * u + 0] = t.x; lq[4 * u + 1] = t.y; lq[4 * u + 2] = t.z; lq[4 * u + 3] = t.w;
    }
  }
  const int* idp = nidx + (size_t)(b * n + col) * k;

  float dots[16];
  float mx = -3.4e38f;
  for (int j = 0; j < k; ++j) {
    int nb = idp[j];
    const float4* k4 = (const float4*)(q2v2 + (size_t)(b * n + nb) * 96 + g * 8);
    float d = 0.f;
#pragma unroll
    for (int u = 0; u < 2; ++u) {
      float4 t = k4[u];
      d += lq[4 * u + 0] * t.x + lq[4 * u + 1] * t.y + lq[4 * u + 2] * t.z + lq[4 * u + 3] * t.w;
    }
    dots[j] = d;
    mx = fmaxf(mx, d);
  }
  float ssum = 0.f;
  for (int j = 0; j < k; ++j) { dots[j] = __expf(dots[j] - mx); ssum += dots[j]; }
  float inv = 1.f / ssum;

  float lo[8];
#pragma unroll
  for (int c = 0; c < 8; ++c) lo[c] = 0.f;
  for (int j = 0; j < k; ++j) {
    int nb = idp[j];
    float la = dots[j] * inv;
    float4 lv = *(const float4*)(q2v2 + (size_t)(b * n + nb) * 96 + 64 + g * 4);
    lo[0] += la * lv.x; lo[1] += la * lv.y; lo[2] += la * lv.z; lo[3] += la * lv.w;
    float p0 = xyz[(size_t)(b * n + nb) * 3 + 0];
    float p1 = xyz[(size_t)(b * n + nb) * 3 + 1];
    float p2 = xyz[(size_t)(b * n + nb) * 3 + 2];
    float r0 = xn0 - p0, r1 = xn1 - p1, r2 = xn2 - p2;
    float dist = sqrtf(r0 * r0 + r1 * r1 + r2 * r2);
    float rp[10] = {dist, r0, r1, r2, xn0, xn1, xn2, p0, p1, p2};
    float fx[8];
#pragma unroll
    for (int c = 0; c < 8; ++c) {
      float s = bl1[c];
#pragma unroll
      for (int t = 0; t < 10; ++t) s += Wl1[c * 10 + t] * rp[t];
      fx[c] = fmaxf(s, 0.f);
    }
#pragma unroll
    for (int c = 0; c < 4; ++c) {
      float s = bl2[c];
#pragma unroll
      for (int t = 0; t < 8; ++t) s += Wl2[c * 8 + t] * fx[t];
      lo[4 + c] += la * fmaxf(s, 0.f);
    }
  }

  // ---- global branch ----
  float gq[8];
  {
    const float4* q4 = (const float4*)(P + 32 + g * 8);
#pragma unroll
    for (int u = 0; u < 2; ++u) {
      float4 t = q4[u];
      gq[4 * u + 0] = t.x; gq[4 * u + 1] = t.y; gq[4 * u + 2] = t.z; gq[4 * u + 3] = t.w;
    }
  }
  float gd[16];
  mx = -3.4e38f;
#pragma unroll
  for (int j = 0; j < 16; ++j) {
    float d = 0.f;
#pragma unroll
    for (int c = 0; c < 8; ++c) d += gq[c] * sgk[j * 8 + c];
    gd[j] = d;
    mx = fmaxf(mx, d);
  }
  ssum = 0.f;
#pragma unroll
  for (int j = 0; j < 16; ++j) { gd[j] = __expf(gd[j] - mx); ssum += gd[j]; }
  inv = 1.f / ssum;

  float go[8];
#pragma unroll
  for (int c = 0; c < 8; ++c) go[c] = 0.f;
#pragma unroll
  for (int j = 0; j < 16; ++j) {
    float ga = gd[j] * inv;
#pragma unroll
    for (int c = 0; c < 4; ++c) go[c] += ga * sgvn[j * 4 + c];
    float p0 = sgxyz[j * 3 + 0], p1 = sgxyz[j * 3 + 1], p2 = sgxyz[j * 3 + 2];
    float r0 = xn0 - p0, r1 = xn1 - p1, r2 = xn2 - p2;
    float dist = sqrtf(r0 * r0 + r1 * r1 + r2 * r2);
    float rp[10] = {dist, r0, r1, r2, xn0, xn1, xn2, p0, p1, p2};
#pragma unroll
    for (int c = 0; c < 4; ++c) {
      float s = bg_[c];
#pragma unroll
      for (int t = 0; t < 10; ++t) s += Wg[c * 10 + t] * rp[t];
      go[4 + c] += ga * fmaxf(s, 0.f);
    }
  }

  _Float16* Ft = f2featT + ((size_t)(b * n + col)) * 96 + g * 16;
#pragma unroll
  for (int c = 0; c < 8; ++c) Ft[c] = (_Float16)lo[c];
#pragma unroll
  for (int c = 0; c < 8; ++c) Ft[8 + c] = (_Float16)go[c];
}

// ---------------------------------------------------------------------------
extern "C" void kernel_launch(void* const* d_in, const int* in_sizes, int n_in,
                              void* d_out, int out_size, void* d_ws, size_t ws_size,
                              hipStream_t stream) {
  const float* feature = (const float*)d_in[0];  // [b,8,n,1]
  const float* xyz     = (const float*)d_in[1];  // [b,n,3]
  const int*   nidx    = (const int*)d_in[2];    // [b,n,k]
  const int B = 2;
  const int N = in_sizes[0] / (8 * B);
  const int K = in_sizes[2] / (B * N);

  // params flattened as JAX pytree (dict keys sorted alphabetically per level):
  // gpe{be,g,m,v,w} lpe1{...} lpe2{...} mlp2{...} qk1{b,w} qk2{b,w} sc{...} v1{...} v2{...}
  int i = 3;
  const float* gpe_be = (const float*)d_in[i++];
  const float* gpe_g  = (const float*)d_in[i++];
  const float* gpe_m  = (const float*)d_in[i++];
  const float* gpe_v  = (const float*)d_in[i++];
  const float* gpe_w  = (const float*)d_in[i++];
  const float* lpe1_be = (const float*)d_in[i++];
  const float* lpe1_g  = (const float*)d_in[i++];
  const float* lpe1_m  = (const float*)d_in[i++];
  const float* lpe1_v  = (const float*)d_in[i++];
  const float* lpe1_w  = (const float*)d_in[i++];
  const float* lpe2_be = (const float*)d_in[i++];
  const float* lpe2_g  = (const float*)d_in[i++];
  const float* lpe2_m  = (const float*)d_in[i++];
  const float* lpe2_v  = (const float*)d_in[i++];
  const float* lpe2_w  = (const float*)d_in[i++];
  const float* mlp2_be = (const float*)d_in[i++];
  const float* mlp2_g  = (const float*)d_in[i++];
  const float* mlp2_m  = (const float*)d_in[i++];
  const float* mlp2_v  = (const float*)d_in[i++];
  const float* mlp2_w  = (const float*)d_in[i++];
  const float* qk1_b = (const float*)d_in[i++];
  const float* qk1_w = (const float*)d_in[i++];
  const float* qk2_b = (const float*)d_in[i++];
  const float* qk2_w = (const float*)d_in[i++];
  const float* sc_be = (const float*)d_in[i++];
  const float* sc_g  = (const float*)d_in[i++];
  const float* sc_m  = (const float*)d_in[i++];
  const float* sc_v  = (const float*)d_in[i++];
  const float* sc_w  = (const float*)d_in[i++];
  const float* v1_be = (const float*)d_in[i++];
  const float* v1_g  = (const float*)d_in[i++];
  const float* v1_m  = (const float*)d_in[i++];
  const float* v1_v  = (const float*)d_in[i++];
  const float* v1_w  = (const float*)d_in[i++];
  const float* v2_be = (const float*)d_in[i++];
  const float* v2_g  = (const float*)d_in[i++];
  const float* v2_m  = (const float*)d_in[i++];
  const float* v2_v  = (const float*)d_in[i++];
  const float* v2_w  = (const float*)d_in[i++];

  // workspace layout
  char* ws = (char*)d_ws;
  size_t off = 0;
  auto alloc = [&](size_t bytes) -> void* {
    void* p = ws + off;
    off = (off + bytes + 255) & ~(size_t)255;
    return p;
  };
  _Float16* A1swz = (_Float16*)alloc(6 * 1 * 512 * sizeof(_Float16));
  _Float16* A2swz = (_Float16*)alloc(6 * 2 * 512 * sizeof(_Float16));
  _Float16* A3swz = (_Float16*)alloc(8 * 3 * 512 * sizeof(_Float16));
  float* bias1 = (float*)alloc(96 * sizeof(float));
  float* bias2 = (float*)alloc(96 * sizeof(float));
  float* bias3 = (float*)alloc(128 * sizeof(float));
  float* Wl1 = (float*)alloc(80 * sizeof(float));
  float* bl1 = (float*)alloc(8 * sizeof(float));
  float* Wl2 = (float*)alloc(32 * sizeof(float));
  float* bl2 = (float*)alloc(4 * sizeof(float));
  float* Wg  = (float*)alloc(40 * sizeof(float));
  float* bg  = (float*)alloc(4 * sizeof(float));
  _Float16* featT   = (_Float16*)alloc((size_t)B * N * 32 * sizeof(_Float16));
  _Float16* f1t     = (_Float16*)alloc((size_t)B * N * 64 * sizeof(_Float16));
  _Float16* f2featT = (_Float16*)alloc((size_t)B * N * 96 * sizeof(_Float16));
  float* q1v1 = (float*)alloc((size_t)B * N * 96 * sizeof(float));  // point-major
  float* q2v2 = (float*)alloc((size_t)B * N * 96 * sizeof(float));  // point-major
  float* ac   = (float*)alloc((size_t)B * 4 * N * sizeof(float));
  float* acs  = (float*)alloc((size_t)B * 4 * N * sizeof(float));
  int*   gidx = (int*)alloc((size_t)B * 4 * 16 * sizeof(int));

  SetupParams sp;
  sp.gpe_be = gpe_be; sp.gpe_g = gpe_g; sp.gpe_m = gpe_m; sp.gpe_v = gpe_v; sp.gpe_w = gpe_w;
  sp.lpe1_be = lpe1_be; sp.lpe1_g = lpe1_g; sp.lpe1_m = lpe1_m; sp.lpe1_v = lpe1_v; sp.lpe1_w = lpe1_w;
  sp.lpe2_be = lpe2_be; sp.lpe2_g = lpe2_g; sp.lpe2_m = lpe2_m; sp.lpe2_v = lpe2_v; sp.lpe2_w = lpe2_w;
  sp.mlp2_be = mlp2_be; sp.mlp2_g = mlp2_g; sp.mlp2_m = mlp2_m; sp.mlp2_v = mlp2_v; sp.mlp2_w = mlp2_w;
  sp.qk1_b = qk1_b; sp.qk1_w = qk1_w; sp.qk2_b = qk2_b; sp.qk2_w = qk2_w;
  sp.sc_be = sc_be; sp.sc_g = sc_g; sp.sc_m = sc_m; sp.sc_v = sc_v; sp.sc_w = sc_w;
  sp.v1_be = v1_be; sp.v1_g = v1_g; sp.v1_m = v1_m; sp.v1_v = v1_v; sp.v1_w = v1_w;
  sp.v2_be = v2_be; sp.v2_g = v2_g; sp.v2_m = v2_m; sp.v2_v = v2_v; sp.v2_w = v2_w;
  sp.A1swz = A1swz; sp.A2swz = A2swz; sp.A3swz = A3swz;
  sp.bias1 = bias1; sp.bias2 = bias2; sp.bias3 = bias3;
  sp.Wl1 = Wl1; sp.bl1 = bl1; sp.Wl2 = Wl2; sp.bl2 = bl2; sp.Wg = Wg; sp.bg = bg;

  setup_kernel<<<1, 256, 0, stream>>>(sp);

  const int acCount = B * 4 * N;
  zero_f32_kernel<<<(acCount + 255) / 256, 256, 0, stream>>>(ac, acCount);

  pack_feature_kernel<<<(B * N) / 256, 256, 0, stream>>>(feature, featT, f2featT, N);

  // q1 || v1 = relu(A1 * feature + bias1): M=96, KP=32, point-major out
  wmma_gemm_kernel<32, false, true><<<dim3(N / 128, 6, B), 256, 0, stream>>>(
      A1swz, bias1, featT, q1v1, N, 96, (long long)N * 32, (long long)N * 96);

  // GTL: local attention + centrality; writes f1t (K-major f16)
  gtl_kernel<<<(B * 4 * N) / 256, 256, 0, stream>>>(q1v1, xyz, nidx, Wl1, bl1, f1t, ac, N, K);

  // q2 || v2 = relu(A2 * f1 + bias2): M=96, KP=64, point-major out
  wmma_gemm_kernel<64, false, true><<<dim3(N / 128, 6, B), 256, 0, stream>>>(
      A2swz, bias2, f1t, q2v2, N, 96, (long long)N * 64, (long long)N * 96);

  // top-16 central points per (b,g)
  topk_kernel<<<B * 4, 256, 0, stream>>>(ac, acs, gidx, N);

  // GT: local + global attention; writes f2featT cols 0..63
  gt_kernel<<<(B * 4 * N) / 256, 256, 0, stream>>>(
      q2v2, xyz, nidx, gidx, Wl1, bl1, Wl2, bl2, Wg, bg, f2featT, N, K);

  // out = leaky_relu(A3 * [f2; feature] + bias3): M=128, KP=96, ref layout out
  wmma_gemm_kernel<96, true, false><<<dim3(N / 128, 8, B), 256, 0, stream>>>(
      A3swz, bias3, f2featT, (float*)d_out, N, 128, (long long)N * 96, (long long)128 * N);
}
